// VanillaLSTM_55362128445951
// MI455X (gfx1250) — matrix operand order repaired and checked by reading, not compiled, not run on hardware
//
#include <hip/hip_runtime.h>
#include <hip/hip_bf16.h>

typedef __attribute__((ext_vector_type(16))) _Float16 v16h;
typedef __attribute__((ext_vector_type(8)))  float    v8f;

#define OBS_LEN 20
#define EDIM 64
#define HDIM 64

union H16 { v16h v; unsigned u[8]; };
union P2  { _Float16 h[2]; unsigned u; };

__device__ __forceinline__ unsigned pack2(float a, float b) {
  P2 p; p.h[0] = (_Float16)a; p.h[1] = (_Float16)b; return p.u;
}
// Native CDNA5 V_TANH_F32 (TRANS op): 1 trans instruction per call.
__device__ __forceinline__ float fast_tanh(float x) {
  return __builtin_amdgcn_tanhf(x);
}
__device__ __forceinline__ float fast_sigmoid(float x) {
  return 0.5f * __builtin_amdgcn_tanhf(0.5f * x) + 0.5f;
}

// Stage a [256][64] f32 matrix into LDS as f16 WMMA A-fragments.
// Fragment (mt,kc) holds gate rows mt*16..mt*16+15, K = kc*32..kc*32+31,
// stored as [lane][16 halves], lane half h holding K offsets 16h..16h+15 (contiguous k order, as in B).
__device__ __forceinline__ void stage_matrix(const float* __restrict__ src,
                                             _Float16* __restrict__ dst, int tid) {
  for (int idx = tid; idx < 16384; idx += 256) {
    int within = idx & 511;
    int lane = within >> 4;
    int j = within & 15;
    int frag = idx >> 9;              // mt*2 + kc
    int mt = frag >> 1, kc = frag & 1;
    int M = mt * 16 + (lane & 15);
    int K = kc * 32 + ((lane & 16) ? 16 : 0) + j;
    dst[idx] = (_Float16)src[M * 64 + K];
  }
}

// Embed (px,py) -> relu(p @ W_emb^T + b_emb) directly into B-fragment halves.
// B layout (32x16 f16): lanes 0-15 hold K=0..15, lanes 16-31 hold K=16..31.
__device__ __forceinline__ void embed(float px, float py,
                                      const float* __restrict__ wemb,
                                      H16 bx[2], int hiK) {
#pragma unroll
  for (int f = 0; f < 2; ++f) {
#pragma unroll
    for (int jj = 0; jj < 8; ++jj) {
      int e0 = (f * 32 + hiK + 2 * jj) * 3;
      float a0 = px * wemb[e0 + 0] + py * wemb[e0 + 1] + wemb[e0 + 2];
      float a1 = px * wemb[e0 + 3] + py * wemb[e0 + 4] + wemb[e0 + 5];
      bx[f].u[jj] = pack2(fmaxf(a0, 0.0f), fmaxf(a1, 0.0f));
    }
  }
}

// One LSTM step for 32 batch columns (two 16-wide B tiles) held by this wave.
// Each A-fragment load now feeds 2 WMMAs -> halves LDS traffic per FLOP.
__device__ __forceinline__ void lstm_step(const _Float16* __restrict__ wfrag,
                                          int mIh, int mHh,
                                          const float* __restrict__ bias,
                                          const H16 bx[2][2], H16 bh[2][2],
                                          float cst[2][4][8], float hall[2][4][8],
                                          int lane) {
  const int hi8 = (lane & 16) ? 8 : 0;
  unsigned pk[2][4][4];
#pragma unroll
  for (int nt = 0; nt < 4; ++nt) {
    v8f acc[4][2];
#pragma unroll
    for (int gi = 0; gi < 4; ++gi) {
      int mt = gi * 4 + nt;                       // tile of gate rows
      v8f binit = *(const v8f*)(bias + gi * 64 + nt * 16 + hi8);
      v16h aih0 = *(const v16h*)(wfrag + ((mIh * 32 + mt * 2 + 0) << 9) + lane * 16);
      v16h aih1 = *(const v16h*)(wfrag + ((mIh * 32 + mt * 2 + 1) << 9) + lane * 16);
      v16h ahh0 = *(const v16h*)(wfrag + ((mHh * 32 + mt * 2 + 0) << 9) + lane * 16);
      v16h ahh1 = *(const v16h*)(wfrag + ((mHh * 32 + mt * 2 + 1) << 9) + lane * 16);
#pragma unroll
      for (int ct = 0; ct < 2; ++ct) {
        v8f a = binit;
        a = __builtin_amdgcn_wmma_f32_16x16x32_f16(false, aih0, false, bx[ct][0].v,
                                                   (short)0, a, false, false);
        a = __builtin_amdgcn_wmma_f32_16x16x32_f16(false, aih1, false, bx[ct][1].v,
                                                   (short)0, a, false, false);
        a = __builtin_amdgcn_wmma_f32_16x16x32_f16(false, ahh0, false, bh[ct][0].v,
                                                   (short)0, a, false, false);
        a = __builtin_amdgcn_wmma_f32_16x16x32_f16(false, ahh1, false, bh[ct][1].v,
                                                   (short)0, a, false, false);
        acc[gi][ct] = a;
      }
    }
    // Elementwise LSTM update: lane holds i,f,g,o for hidden n = nt*16+hi8+r
#pragma unroll
    for (int ct = 0; ct < 2; ++ct) {
#pragma unroll
      for (int r = 0; r < 8; ++r) {
        float ig = fast_sigmoid(acc[0][ct][r]);
        float fg = fast_sigmoid(acc[1][ct][r]);
        float gg = fast_tanh(acc[2][ct][r]);
        float og = fast_sigmoid(acc[3][ct][r]);
        float c2 = fg * cst[ct][nt][r] + ig * gg;
        cst[ct][nt][r] = c2;
        hall[ct][nt][r] = og * fast_tanh(c2);
      }
#pragma unroll
      for (int r = 0; r < 4; ++r)
        pk[ct][nt][r] = pack2(hall[ct][nt][2 * r], hall[ct][nt][2 * r + 1]);
    }
  }
  // Rebuild h as B-fragments: one xor-16 lane swap per fragment half.
  const bool low = (lane & 16) == 0;
#pragma unroll
  for (int ct = 0; ct < 2; ++ct)
#pragma unroll
    for (int f = 0; f < 2; ++f)
#pragma unroll
      for (int w = 0; w < 4; ++w) {
        unsigned q = low ? pk[ct][2 * f + 1][w] : pk[ct][2 * f][w];
        unsigned s = (unsigned)__shfl_xor((int)q, 16, 32);
        if (low) { bh[ct][f].u[w] = pk[ct][2 * f][w]; bh[ct][f].u[4 + w] = s; }
        else     { bh[ct][f].u[w] = s;                bh[ct][f].u[4 + w] = pk[ct][2 * f + 1][w]; }
      }
}

extern __shared__ char smem[];

__global__ void __launch_bounds__(256)
vanilla_lstm_kernel(const float* __restrict__ obs,
                    const float* __restrict__ Wemb, const float* __restrict__ bemb,
                    const float* __restrict__ WihE, const float* __restrict__ WhhE,
                    const float* __restrict__ bihE, const float* __restrict__ bhhE,
                    const float* __restrict__ WihD, const float* __restrict__ WhhD,
                    const float* __restrict__ bihD, const float* __restrict__ bhhD,
                    const float* __restrict__ Wfc, const float* __restrict__ bfc,
                    float* __restrict__ out, int predLen) {
  _Float16* wfrag = (_Float16*)smem;                 // 4 * 16384 halves = 128KB
  float* be   = (float*)(smem + 131072);             // 256
  float* bd   = be + 256;                            // 256
  float* wemb = bd + 256;                            // 64*3 interleaved (w0,w1,b)
  float* wfc  = wemb + 192;                          // 128
  float* bfcs = wfc + 128;                           // 2

  const int tid = threadIdx.x;
  stage_matrix(WihE, wfrag + 0 * 16384, tid);
  stage_matrix(WhhE, wfrag + 1 * 16384, tid);
  stage_matrix(WihD, wfrag + 2 * 16384, tid);
  stage_matrix(WhhD, wfrag + 3 * 16384, tid);
  if (tid < 256) { be[tid] = bihE[tid] + bhhE[tid]; bd[tid] = bihD[tid] + bhhD[tid]; }
  if (tid < 64) {
    wemb[tid * 3 + 0] = Wemb[tid * 2 + 0];
    wemb[tid * 3 + 1] = Wemb[tid * 2 + 1];
    wemb[tid * 3 + 2] = bemb[tid];
  }
  if (tid < 128) wfc[tid] = Wfc[tid];
  if (tid < 2)  bfcs[tid] = bfc[tid];
  __syncthreads();

  const int lane  = tid & 31;
  const int wave  = tid >> 5;
  const int gwave = blockIdx.x * 8 + wave;
  const int colBase = gwave * 32;                    // 32 batch rows per wave
  const int row0 = colBase + (lane & 15);            // tile ct=0
  const int row1 = row0 + 16;                        // tile ct=1
  const int hi8 = (lane & 16) ? 8 : 0;
  const int hiK = (lane & 16) ? 16 : 0;

  float cst[2][4][8], hall[2][4][8];
  H16 bx[2][2], bh[2][2];
#pragma unroll
  for (int ct = 0; ct < 2; ++ct) {
#pragma unroll
    for (int nt = 0; nt < 4; ++nt)
#pragma unroll
      for (int r = 0; r < 8; ++r) { cst[ct][nt][r] = 0.0f; hall[ct][nt][r] = 0.0f; }
#pragma unroll
    for (int f = 0; f < 2; ++f)
#pragma unroll
      for (int w = 0; w < 8; ++w) bh[ct][f].u[w] = 0u;
  }

  // ---------------- Encoder ----------------
  const float2* obs2 = (const float2*)obs;
#pragma clang loop unroll(disable)
  for (int t = 0; t < OBS_LEN; ++t) {
    // Block LICM from hoisting (and then spilling) the LDS weight-fragment
    // loads out of the time loop: force them to stay as in-loop ds_loads.
    asm volatile("" ::: "memory");
    float2 o0 = obs2[(size_t)row0 * OBS_LEN + t];
    float2 o1 = obs2[(size_t)row1 * OBS_LEN + t];
    embed(o0.x, o0.y, wemb, bx[0], hiK);
    embed(o1.x, o1.y, wemb, bx[1], hiK);
    lstm_step(wfrag, 0, 1, be, bx, bh, cst, hall, lane);
  }

  // ---------------- Decoder ---------------- (bx already = embed(obs[:, -1]))
#pragma clang loop unroll(disable)
  for (int t = 0; t < predLen; ++t) {
    asm volatile("" ::: "memory");
    lstm_step(wfrag, 2, 3, bd, bx, bh, cst, hall, lane);
#pragma unroll
    for (int ct = 0; ct < 2; ++ct) {
      // pred = h @ W_fc^T + b_fc : lane holds 32 of 64 h-values for its column
      float p0 = 0.0f, p1 = 0.0f;
#pragma unroll
      for (int nt = 0; nt < 4; ++nt)
#pragma unroll
        for (int r = 0; r < 8; ++r) {
          int n = nt * 16 + hi8 + r;
          p0 += hall[ct][nt][r] * wfc[n];
          p1 += hall[ct][nt][r] * wfc[64 + n];
        }
      p0 += __shfl_xor(p0, 16, 32);
      p1 += __shfl_xor(p1, 16, 32);
      p0 += bfcs[0];
      p1 += bfcs[1];
      int rowc = colBase + ct * 16 + (lane & 15);
      if ((lane & 16) == 0) {
        float2* op = (float2*)(out + ((size_t)rowc * predLen + t) * 2);
        *op = make_float2(p0, p1);
      }
      embed(p0, p1, wemb, bx[ct], hiK);   // next decoder input
    }
  }
}

extern "C" void kernel_launch(void* const* d_in, const int* in_sizes, int n_in,
                              void* d_out, int out_size, void* d_ws, size_t ws_size,
                              hipStream_t stream) {
  const float* obs  = (const float*)d_in[0];
  const float* Wemb = (const float*)d_in[1];
  const float* bemb = (const float*)d_in[2];
  const float* WihE = (const float*)d_in[3];
  const float* WhhE = (const float*)d_in[4];
  const float* bihE = (const float*)d_in[5];
  const float* bhhE = (const float*)d_in[6];
  const float* WihD = (const float*)d_in[7];
  const float* WhhD = (const float*)d_in[8];
  const float* bihD = (const float*)d_in[9];
  const float* bhhD = (const float*)d_in[10];
  const float* Wfc  = (const float*)d_in[11];
  const float* bfc  = (const float*)d_in[12];
  float* out = (float*)d_out;

  const int B = in_sizes[0] / (OBS_LEN * 2);
  const int predLen = out_size / (B * 2);
  const int blocks = B / 256;              // 8 waves * 32 batch rows per block
  const size_t shmem = 131072 + (256 + 256 + 192 + 128 + 2) * sizeof(float) + 64;

  vanilla_lstm_kernel<<<blocks, 256, shmem, stream>>>(
      obs, Wemb, bemb, WihE, WhhE, bihE, bhhE,
      WihD, WhhD, bihD, bhhD, Wfc, bfc, out, predLen);
}
